// ETFGraphModel_13194139533588
// MI455X (gfx1250) — matrix operand order repaired and checked by reading, not compiled, not run on hardware
//
#include <hip/hip_runtime.h>
#include <hip/hip_bf16.h>
#include <math.h>

#define N_NODES 16384
#define N_EDGES 524288
#define ETOT    (N_EDGES + N_NODES)   /* 540672 */
#define HEADS   3
#define HID     64
#define IN_DIM  128
#define CDIM    (HEADS * HID)         /* 192 */
#define NEG_SLOPE 0.2f

typedef __bf16 bf16_t;
typedef __attribute__((ext_vector_type(16))) __bf16 v16bf;
typedef __attribute__((ext_vector_type(8)))  __bf16 bf16x8;
typedef __attribute__((ext_vector_type(8)))  float  v8f;

// ---------------- helpers ----------------
__device__ __forceinline__ float sigmoidf_(float x) { return 1.f / (1.f + __expf(-x)); }

// monotone float <-> uint encoding for atomicMax on floats
__device__ __forceinline__ unsigned fenc_(float f) {
    unsigned u = __float_as_uint(f);
    return (u & 0x80000000u) ? ~u : (u | 0x80000000u);
}
__device__ __forceinline__ float fdec_(unsigned u) {
    unsigned b = (u & 0x80000000u) ? (u ^ 0x80000000u) : ~u;
    return __uint_as_float(b);
}
#define ENC_NEG_FLT_MAX 0x00800000u   /* fenc_(-FLT_MAX) */

// A fragment (16x32 bf16, MxK): lanes 0-15 row M=lane, halves K0-7 then K16-23;
// lanes 16-31 same rows, K8-15 then K24-31 (ISA 7.12.2).
__device__ __forceinline__ v16bf load_a_frag(const bf16_t* base, int lda, int row0, int k0, int lane) {
    int r = lane & 15, half = lane >> 4;
    const bf16_t* p = base + (long)(row0 + r) * lda + k0;
    bf16x8 lo = *reinterpret_cast<const bf16x8*>(p + half * 8);
    bf16x8 hi = *reinterpret_cast<const bf16x8*>(p + 16 + half * 8);
    v16bf a;
#pragma unroll
    for (int i = 0; i < 8; i++) { a[i] = lo[i]; a[8 + i] = hi[i]; }
    return a;
}
// B fragment (32x16 bf16, KxN), source given row-major as rows = N, cols = K
// (i.e. B = src^T). Lane holds column n = lane&15; half 0 -> K0-7 then K16-23, half 1 -> K8-15 then K24-31.
__device__ __forceinline__ v16bf load_b_frag(const bf16_t* src, int lds_, int col0, int k0, int lane) {
    int n = lane & 15, half = lane >> 4;
    const bf16_t* p = src + (long)(col0 + n) * lds_ + k0 + half * 8;
    bf16x8 lo = *reinterpret_cast<const bf16x8*>(p);
    bf16x8 hi = *reinterpret_cast<const bf16x8*>(p + 16);
    v16bf b;
#pragma unroll
    for (int i = 0; i < 8; i++) { b[i] = lo[i]; b[8 + i] = hi[i]; }
    return b;
}

// ---------------- conversion ----------------
__global__ void f32_to_bf16_kernel(const float* __restrict__ in, bf16_t* __restrict__ out, int n) {
    int i = blockIdx.x * 256 + threadIdx.x;
    if (i < n) out[i] = (bf16_t)in[i];
}

// ---------------- GAT: h = x @ lin_w.T  (WMMA bf16) ----------------
// grid (3, 1024), block 128 (4 waves). wave handles n-tile = bx*4+wave (12 n-tiles), m-tile = by.
__global__ __launch_bounds__(128) void gemm_h_kernel(const bf16_t* __restrict__ xb,
                                                     const bf16_t* __restrict__ wb,
                                                     float* __restrict__ h) {
    int lane = threadIdx.x & 31;
    int wave = threadIdx.x >> 5;
    int tn = blockIdx.x * 4 + wave;   // 0..11
    int tm = blockIdx.y;              // 0..1023
    v8f c = {};
#pragma unroll
    for (int k0 = 0; k0 < IN_DIM; k0 += 32) {
        v16bf a = load_a_frag(xb, IN_DIM, tm * 16, k0, lane);
        v16bf b = load_b_frag(wb, IN_DIM, tn * 16, k0, lane);
        c = __builtin_amdgcn_wmma_f32_16x16x32_bf16(false, a, false, b, (short)0, c, false, false);
    }
    int n = lane & 15, half = lane >> 4;
#pragma unroll
    for (int v = 0; v < 8; v++) {
        int row = tm * 16 + half * 8 + v;
        h[(long)row * CDIM + tn * 16 + n] = c[v];
    }
}

// ---------------- attention scalars ----------------
// grid (64, 3) block 256 : a_src[n,hd], a_dst[n,hd]
__global__ void attn_scalar_kernel(const float* __restrict__ h,
                                   const float* __restrict__ att_src,
                                   const float* __restrict__ att_dst,
                                   float* __restrict__ a_src, float* __restrict__ a_dst) {
    int n = blockIdx.x * 256 + threadIdx.x;
    int hd = blockIdx.y;
    const float* hv = h + (long)n * CDIM + hd * HID;
    const float* ws = att_src + hd * HID;
    const float* wd = att_dst + hd * HID;
    float as = 0.f, ad = 0.f;
#pragma unroll 8
    for (int c = 0; c < HID; c++) { float v = hv[c]; as += v * ws[c]; ad += v * wd[c]; }
    a_src[n * HEADS + hd] = as;
    a_dst[n * HEADS + hd] = ad;
}

// ---------------- init accumulators ----------------
__global__ void init_kernel(unsigned* __restrict__ m_enc, float* __restrict__ z, float* __restrict__ accum) {
    int i = blockIdx.x * 256 + threadIdx.x;
    if (i < N_NODES * CDIM) accum[i] = 0.f;
    if (i < N_NODES * HEADS) { m_enc[i] = ENC_NEG_FLT_MAX; z[i] = 0.f; }
}

__device__ __forceinline__ void edge_sd(const long long* ei, int idx, int& s, int& d) {
    if (idx < N_EDGES) { s = (int)ei[idx]; d = (int)ei[N_EDGES + idx]; }
    else { s = idx - N_EDGES; d = s; }
}

// pass A: segment max (monotone-uint atomicMax)
__global__ void edge_max_kernel(const long long* __restrict__ ei,
                                const float* __restrict__ a_src, const float* __restrict__ a_dst,
                                unsigned* __restrict__ m_enc) {
    int idx = blockIdx.x * 256 + threadIdx.x;
    if (idx >= ETOT) return;
    int s, d; edge_sd(ei, idx, s, d);
#pragma unroll
    for (int hd = 0; hd < HEADS; hd++) {
        float e = a_src[s * HEADS + hd] + a_dst[d * HEADS + hd];
        e = e > 0.f ? e : NEG_SLOPE * e;
        atomicMax(&m_enc[d * HEADS + hd], fenc_(e));
    }
}

// pass B: z = segment sum of exp(e - m)
__global__ void edge_sum_kernel(const long long* __restrict__ ei,
                                const float* __restrict__ a_src, const float* __restrict__ a_dst,
                                const unsigned* __restrict__ m_enc, float* __restrict__ z) {
    int idx = blockIdx.x * 256 + threadIdx.x;
    if (idx >= ETOT) return;
    int s, d; edge_sd(ei, idx, s, d);
#pragma unroll
    for (int hd = 0; hd < HEADS; hd++) {
        float e = a_src[s * HEADS + hd] + a_dst[d * HEADS + hd];
        e = e > 0.f ? e : NEG_SLOPE * e;
        atomicAdd(&z[d * HEADS + hd], __expf(e - fdec_(m_enc[d * HEADS + hd])));
    }
}

// pass C: accum[dst] += h[src] * alpha   (L2-resident scatter)
// grid (2112, 3) block 256, blockIdx.y = head
__global__ void edge_scatter_kernel(const long long* __restrict__ ei,
                                    const float* __restrict__ a_src, const float* __restrict__ a_dst,
                                    const unsigned* __restrict__ m_enc, const float* __restrict__ z,
                                    const float* __restrict__ h, float* __restrict__ accum) {
    int idx = blockIdx.x * 256 + threadIdx.x;
    if (idx >= ETOT) return;
    int hd = blockIdx.y;
    int s, d; edge_sd(ei, idx, s, d);
    float e = a_src[s * HEADS + hd] + a_dst[d * HEADS + hd];
    e = e > 0.f ? e : NEG_SLOPE * e;
    float alpha = __expf(e - fdec_(m_enc[d * HEADS + hd])) / z[d * HEADS + hd];
    const float* hs = h + (long)s * CDIM + hd * HID;
    float* acc = accum + (long)d * CDIM + hd * HID;
#pragma unroll 4
    for (int c = 0; c < HID; c++) atomicAdd(&acc[c], hs[c] * alpha);
}

// g[n,c] = mean over heads + bias
__global__ void finalize_g_kernel(const float* __restrict__ accum, const float* __restrict__ bias,
                                  float* __restrict__ g) {
    int i = blockIdx.x * 256 + threadIdx.x;   // n*64+c
    int n = i >> 6, c = i & 63;
    const float* a = accum + (long)n * CDIM;
    g[i] = (a[c] + a[HID + c] + a[2 * HID + c]) * (1.f / 3.f) + bias[c];
}

// ---------------- LSTM scan: one workgroup, register-resident weights ----------------
__global__ __launch_bounds__(256, 1) void lstm_kernel(const float* __restrict__ g,
                                                      const float* __restrict__ w_ih,
                                                      const float* __restrict__ w_hh,
                                                      const float* __restrict__ b_ih,
                                                      const float* __restrict__ b_hh,
                                                      float* __restrict__ hs) {
    __shared__ float s_x[HID], s_h[HID], s_c[HID], s_g[4 * HID];
    int tid = threadIdx.x;             // gate row 0..255
    float wih[HID], whh[HID];
#pragma unroll
    for (int k = 0; k < HID; k++) { wih[k] = w_ih[tid * HID + k]; whh[k] = w_hh[tid * HID + k]; }
    float bias = b_ih[tid] + b_hh[tid];
    if (tid < HID) { s_h[tid] = 0.f; s_c[tid] = 0.f; }
    __syncthreads();
    for (int t = 0; t < N_NODES; t++) {
        if (tid < HID) s_x[tid] = g[(long)t * HID + tid];
        __syncthreads();
        float acc = bias;
#pragma unroll
        for (int k = 0; k < HID; k++) acc += wih[k] * s_x[k];
#pragma unroll
        for (int k = 0; k < HID; k++) acc += whh[k] * s_h[k];
        s_g[tid] = acc;
        __syncthreads();
        if (tid < HID) {
            float ig = sigmoidf_(s_g[tid]);
            float fg = sigmoidf_(s_g[HID + tid]);
            float gg = tanhf(s_g[2 * HID + tid]);
            float og = sigmoidf_(s_g[3 * HID + tid]);
            float cn = fg * s_c[tid] + ig * gg;
            float hn = og * tanhf(cn);
            s_c[tid] = cn; s_h[tid] = hn;
            hs[(long)t * HID + tid] = hn;
        }
        __syncthreads();
    }
}

// ---------------- normalize rows of h -> bf16 ----------------
__global__ void norm_kernel(const float* __restrict__ h, bf16_t* __restrict__ hn) {
    int n = blockIdx.x * 256 + threadIdx.x;
    const float* r = h + (long)n * HID;
    float ss = 0.f;
#pragma unroll 8
    for (int c = 0; c < HID; c++) ss += r[c] * r[c];
    float inv = 1.f / fmaxf(sqrtf(ss), 1e-12f);
    bf16_t* o = hn + (long)n * HID;
#pragma unroll 8
    for (int c = 0; c < HID; c++) o[c] = (bf16_t)(r[c] * inv);
}

// ---------------- corr = hn @ hn^T  (WMMA bf16, store-bound: 1.07 GB) ----------------
// grid (256, 1024), block 128 (4 waves). wave -> col tile tn = bx*4+wave, row tile tm = by.
__global__ __launch_bounds__(128) void corr_kernel(const bf16_t* __restrict__ hn, float* __restrict__ corr) {
    int lane = threadIdx.x & 31;
    int wave = threadIdx.x >> 5;
    int tn = blockIdx.x * 4 + wave;   // 0..1023
    int tm = blockIdx.y;              // 0..1023
    v8f c = {};
#pragma unroll
    for (int k0 = 0; k0 < HID; k0 += 32) {
        v16bf a = load_a_frag(hn, HID, tm * 16, k0, lane);
        v16bf b = load_b_frag(hn, HID, tn * 16, k0, lane);
        c = __builtin_amdgcn_wmma_f32_16x16x32_bf16(false, a, false, b, (short)0, c, false, false);
    }
    int n = lane & 15, half = lane >> 4;
#pragma unroll
    for (int v = 0; v < 8; v++) {
        long row = tm * 16 + half * 8 + v;
        corr[row * N_NODES + tn * 16 + n] = c[v];
    }
}

// ---------------- uncertainty MLP head ----------------
__global__ __launch_bounds__(256) void mlp_kernel(const float* __restrict__ h,
                                                  const float* __restrict__ u_w1, const float* __restrict__ u_b1,
                                                  const float* __restrict__ u_w2, const float* __restrict__ u_b2,
                                                  float* __restrict__ mu, float* __restrict__ sigma) {
    __shared__ float w1[32 * HID], b1[32], w2[64], b2[2];
    for (int i = threadIdx.x; i < 32 * HID; i += 256) w1[i] = u_w1[i];
    if (threadIdx.x < 32) b1[threadIdx.x] = u_b1[threadIdx.x];
    if (threadIdx.x < 64) w2[threadIdx.x] = u_w2[threadIdx.x];
    if (threadIdx.x < 2) b2[threadIdx.x] = u_b2[threadIdx.x];
    __syncthreads();
    int n = blockIdx.x * 256 + threadIdx.x;
    const float* r = h + (long)n * HID;
    float m = b2[0], s = b2[1];
    for (int j = 0; j < 32; j++) {
        float acc = b1[j];
        const float* w = w1 + j * HID;
#pragma unroll 8
        for (int c = 0; c < HID; c++) acc += w[c] * r[c];
        acc = fmaxf(acc, 0.f);
        m += acc * w2[j];          // u_w2[0][j]
        s += acc * w2[32 + j];     // u_w2[1][j]
    }
    mu[n] = m;
    sigma[n] = s;
}

// ---------------- launch ----------------
extern "C" void kernel_launch(void* const* d_in, const int* in_sizes, int n_in,
                              void* d_out, int out_size, void* d_ws, size_t ws_size,
                              hipStream_t stream) {
    const float*     x       = (const float*)d_in[0];
    const long long* ei      = (const long long*)d_in[1];   // int64 edge_index
    const float*     lin_w   = (const float*)d_in[2];
    const float*     att_src = (const float*)d_in[3];
    const float*     att_dst = (const float*)d_in[4];
    const float*     gat_b   = (const float*)d_in[5];
    const float*     w_ih    = (const float*)d_in[6];
    const float*     w_hh    = (const float*)d_in[7];
    const float*     b_ih    = (const float*)d_in[8];
    const float*     b_hh    = (const float*)d_in[9];
    const float*     u_w1    = (const float*)d_in[10];
    const float*     u_b1    = (const float*)d_in[11];
    const float*     u_w2    = (const float*)d_in[12];
    const float*     u_b2    = (const float*)d_in[13];

    float* out = (float*)d_out;
    float* out_h    = out;                                         // N*64
    float* out_corr = out + (long)N_NODES * HID;                   // N*N
    float* out_mu   = out_corr + (long)N_NODES * N_NODES;          // N
    float* out_sig  = out_mu + N_NODES;                            // N

    // workspace carve-up
    char* ws = (char*)d_ws;
    size_t off = 0;
    auto take = [&](size_t bytes) { void* p = ws + off; off += (bytes + 255) & ~(size_t)255; return p; };
    float*    h_gat  = (float*)take((size_t)N_NODES * CDIM * 4);   // 12.6 MB
    float*    a_src  = (float*)take((size_t)N_NODES * HEADS * 4);
    float*    a_dst  = (float*)take((size_t)N_NODES * HEADS * 4);
    unsigned* m_enc  = (unsigned*)take((size_t)N_NODES * HEADS * 4);
    float*    z      = (float*)take((size_t)N_NODES * HEADS * 4);
    float*    accum  = (float*)take((size_t)N_NODES * CDIM * 4);   // 12.6 MB
    float*    g      = (float*)take((size_t)N_NODES * HID * 4);    // 4.2 MB
    bf16_t*   x_bf   = (bf16_t*)take((size_t)N_NODES * IN_DIM * 2);
    bf16_t*   w_bf   = (bf16_t*)take((size_t)CDIM * IN_DIM * 2);
    bf16_t*   hn_bf  = (bf16_t*)take((size_t)N_NODES * HID * 2);
    (void)ws_size; (void)in_sizes; (void)n_in; (void)out_size;

    // 1. bf16 conversions for WMMA operands
    f32_to_bf16_kernel<<<(N_NODES * IN_DIM) / 256, 256, 0, stream>>>(x, x_bf, N_NODES * IN_DIM);
    f32_to_bf16_kernel<<<(CDIM * IN_DIM + 255) / 256, 256, 0, stream>>>(lin_w, w_bf, CDIM * IN_DIM);

    // 2. h = x @ lin_w.T  (WMMA)
    gemm_h_kernel<<<dim3(3, 1024), 128, 0, stream>>>(x_bf, w_bf, h_gat);

    // 3. attention scalars
    attn_scalar_kernel<<<dim3(N_NODES / 256, HEADS), 256, 0, stream>>>(h_gat, att_src, att_dst, a_src, a_dst);

    // 4. init + edge softmax + scatter
    init_kernel<<<(N_NODES * CDIM) / 256, 256, 0, stream>>>(m_enc, z, accum);
    edge_max_kernel<<<ETOT / 256, 256, 0, stream>>>(ei, a_src, a_dst, m_enc);
    edge_sum_kernel<<<ETOT / 256, 256, 0, stream>>>(ei, a_src, a_dst, m_enc, z);
    edge_scatter_kernel<<<dim3(ETOT / 256, HEADS), 256, 0, stream>>>(ei, a_src, a_dst, m_enc, z, h_gat, accum);
    finalize_g_kernel<<<(N_NODES * HID) / 256, 256, 0, stream>>>(accum, gat_b, g);

    // 5. sequential LSTM scan -> out_h
    lstm_kernel<<<1, 256, 0, stream>>>(g, w_ih, w_hh, b_ih, b_hh, out_h);

    // 6. row-normalize -> bf16
    norm_kernel<<<N_NODES / 256, 256, 0, stream>>>(out_h, hn_bf);

    // 7. corr = hn @ hn^T (WMMA, store-bound)
    corr_kernel<<<dim3(256, 1024), 128, 0, stream>>>(hn_bf, out_corr);

    // 8. uncertainty head
    mlp_kernel<<<N_NODES / 256, 256, 0, stream>>>(out_h, u_w1, u_b1, u_w2, u_b2, out_mu, out_sig);
}
